// GAT_10282151706762
// MI455X (gfx1250) — matrix operand order repaired and checked by reading, not compiled, not run on hardware
//
#include <hip/hip_runtime.h>

// GAT (2x GATConv + GlobalAttention pool) for gfx1250.
// Dense GEMMs: v_wmma_f32_16x16x32_f16, f32 accum. Weights pre-swizzled into
// WMMA-fragment-major order so B fragments load as 2x global_load_b128.
// Each wave owns 16x(NC/4) of C (NC/64 accumulators) so A fragments are loaded once
// per block. Edge softmax/scatter: ordered-float atomicMax (uint flip) +
// f32 atomicAdd; h-tables (102MB) are L2-resident on MI455X (192MB L2).
// Workspace use: ~345 MB.

typedef __attribute__((ext_vector_type(16))) _Float16 v16h;
typedef __attribute__((ext_vector_type(8)))  float    v8f;

__device__ __forceinline__ unsigned fflip(float f) {
  unsigned u = __float_as_uint(f);
  return (u & 0x80000000u) ? ~u : (u | 0x80000000u);
}
__device__ __forceinline__ float funflip(unsigned u) {
  return __uint_as_float((u & 0x80000000u) ? (u & 0x7FFFFFFFu) : ~u);
}
__device__ __forceinline__ float sigmoidf_(float x) { return 1.0f / (1.0f + expf(-x)); }

// ---------------------------------------------------------------- fills
__global__ void fill_u32_kernel(unsigned* __restrict__ p, unsigned v, size_t n) {
  size_t i = (size_t)blockIdx.x * blockDim.x + threadIdx.x;
  size_t stride = (size_t)gridDim.x * blockDim.x;
  for (; i < n; i += stride) p[i] = v;
}

// ----------------------- weight pack: f32[K,NC] -> f16 fragment-major uints
// out index: (((bn*(K/32) + ks)*32 + lane)*8 + r)
//   lane: n = bn*16 + (lane&15), K-half = lane>>4
//   r:    kb = ks*32 + 16*(lane>>4) + 2r ; value packs (W[kb][n], W[kb+1][n])
__global__ void pack_w_frag_kernel(const float* __restrict__ W, unsigned* __restrict__ out,
                                   int K, int NC) {
  int i = blockIdx.x * blockDim.x + threadIdx.x;
  int total = (K >> 1) * NC;
  if (i >= total) return;
  int r = i & 7;
  int l = (i >> 3) & 31;
  int rest = i >> 8;
  int ksteps = K >> 5;
  int ks = rest % ksteps;
  int bn = rest / ksteps;
  int col = bn * 16 + (l & 15);
  int kb = ks * 32 + ((l >> 4) * 16) + 2 * r;
  union { unsigned u; _Float16 h[2]; } pk;
  pk.h[0] = (_Float16)W[(size_t)kb * NC + col];
  pk.h[1] = (_Float16)W[(size_t)(kb + 1) * NC + col];
  out[i] = pk.u;
}

// ------------------------------------------- embedding gather -> f16 [N,384]
__global__ void embed_kernel(const int* __restrict__ xi,
                             const float* e0, const float* e1, const float* e2,
                             const float* e3, const float* e4, const float* e5,
                             _Float16* __restrict__ out, int n) {
  int i = blockIdx.x * blockDim.x + threadIdx.x;
  if (i >= n * 384) return;
  int node = i / 384, j = i % 384;
  int t = j >> 6, w = j & 63;
  const float* tabs[6] = {e0, e1, e2, e3, e4, e5};
  int idx = xi[node * 6 + t];
  out[i] = (_Float16)tabs[t][(size_t)idx * 64 + w];
}

// ------------------------------------------- WMMA GEMM: C[N,NC] = A[N,K]f16 * Bfrag
// Block = 4 waves, one 16-row tile. Wave w owns cols [w*NC/4, (w+1)*NC/4) via
// TPW = NC/64 accumulators; A fragment loaded once per k-step per wave.
template <int K, int NC>
__global__ __launch_bounds__(128)
void gemm_wmma_kernel(const unsigned* __restrict__ A, const uint4* __restrict__ Bp,
                      float* __restrict__ C, int nrows) {
  constexpr int TPW = NC / 64;   // 16-col tiles per wave
  constexpr int KS = K / 32;
  int lane = threadIdx.x & 31;
  int wave = threadIdx.x >> 5;
  int row_base = blockIdx.x * 16;
  int hg = lane >> 4;            // lane-half selects K-subgroup
  int m = lane & 15;
  int row = row_base + m;
  int rowc = row < nrows ? row : nrows - 1; // clamp loads; EXEC stays full
  const int kpr = K >> 1;

  v8f acc[TPW];
#pragma unroll
  for (int t = 0; t < TPW; ++t) acc[t] = (v8f){};

#pragma unroll
  for (int ks = 0; ks < KS; ++ks) {
    union { v16h v; unsigned u[8]; } a;
#pragma unroll
    for (int r = 0; r < 8; ++r) {
      int ka = ks * 32 + 16 * hg + 2 * r;
      a.u[r] = A[(size_t)rowc * kpr + (ka >> 1)];   // contiguous -> b128 merges
    }
#pragma unroll
    for (int t = 0; t < TPW; ++t) {
      int bn = wave * TPW + t;
      union { v16h v; uint4 q[2]; } b;
      size_t fidx = (((size_t)(bn * KS + ks)) * 32 + lane) * 2; // uint4 units
      b.q[0] = Bp[fidx];
      b.q[1] = Bp[fidx + 1];
      acc[t] = __builtin_amdgcn_wmma_f32_16x16x32_f16(false, a.v, false, b.v,
                                                      (short)0, acc[t], false, false);
    }
  }
#pragma unroll
  for (int t = 0; t < TPW; ++t) {
    int col = (wave * TPW + t) * 16 + m;
#pragma unroll
    for (int r = 0; r < 8; ++r) {
      int mr = row_base + r + 8 * hg;
      if (mr < nrows) C[(size_t)mr * NC + col] = acc[t][r];
    }
  }
}

// ------------------------------------------- attention logits: wave per (node,head)
__global__ void attn_logits_kernel(const float* __restrict__ H,
                                   const float* __restrict__ asrc,
                                   const float* __restrict__ adst,
                                   float* __restrict__ als, float* __restrict__ ald,
                                   int n) {
  int w = blockIdx.x * (blockDim.x >> 5) + (threadIdx.x >> 5);
  if (w >= n * 2) return;
  int node = w >> 1, h = w & 1;
  int lane = threadIdx.x & 31;
  const float* hp = H + (size_t)node * 256 + h * 128;
  float ss = 0.f, sd = 0.f;
#pragma unroll
  for (int i = 0; i < 4; ++i) {
    int c = lane + 32 * i;
    float v = hp[c];
    ss += v * asrc[h * 128 + c];
    sd += v * adst[h * 128 + c];
  }
#pragma unroll
  for (int o = 16; o; o >>= 1) { ss += __shfl_xor(ss, o, 32); sd += __shfl_xor(sd, o, 32); }
  if (lane == 0) { als[w] = ss; ald[w] = sd; }
}

// ------------------------------------------- edge pass 1: leaky_relu + seg max
__global__ void edge_max_kernel(const int* __restrict__ src, const int* __restrict__ dst,
                                const float* __restrict__ als, const float* __restrict__ ald,
                                float* __restrict__ ew, unsigned* __restrict__ maxb,
                                int E, int ET) {
  int e = blockIdx.x * blockDim.x + threadIdx.x;
  if (e >= ET) return;
  int s, d;
  if (e < E) { s = src[e]; d = dst[e]; } else { s = d = e - E; }
#pragma unroll
  for (int h = 0; h < 2; ++h) {
    float l = als[s * 2 + h] + ald[d * 2 + h];
    l = l > 0.f ? l : 0.2f * l;
    ew[(size_t)e * 2 + h] = l;
    atomicMax(&maxb[d * 2 + h], fflip(l));
  }
}

// ------------------------------------------- edge pass 2: exp + seg sum
__global__ void edge_exp_kernel(const int* __restrict__ dst_arr,
                                float* __restrict__ ew, const unsigned* __restrict__ maxb,
                                float* __restrict__ sumb, int E, int ET) {
  int e = blockIdx.x * blockDim.x + threadIdx.x;
  if (e >= ET) return;
  int d = (e < E) ? dst_arr[e] : e - E;
#pragma unroll
  for (int h = 0; h < 2; ++h) {
    float ex = expf(ew[(size_t)e * 2 + h] - funflip(maxb[d * 2 + h]));
    ew[(size_t)e * 2 + h] = ex;
    atomicAdd(&sumb[d * 2 + h], ex);
  }
}

// -------------------------- edge pass 3: alpha * h[src] -> acc[dst], 32 thr/edge
__global__ void edge_scatter_kernel(const int* __restrict__ src, const int* __restrict__ dst,
                                    const float* __restrict__ ew, const float* __restrict__ sumb,
                                    const float* __restrict__ H, float* __restrict__ acc,
                                    int E, int ET) {
  long tid = (long)blockIdx.x * blockDim.x + threadIdx.x;
  long total = (long)ET * 32;
  if (tid >= total) return;
  int e = (int)(tid >> 5);
  int q = (int)(tid & 31);
  int s, d;
  if (e < E) { s = src[e]; d = dst[e]; } else { s = d = e - E; }
#pragma unroll
  for (int h = 0; h < 2; ++h) {
    float alpha = ew[(size_t)e * 2 + h] / (sumb[d * 2 + h] + 1e-16f);
    float4 hv = *(const float4*)(H + (size_t)s * 256 + h * 128 + 4 * q); // b128 gather
    float* ap = acc + (size_t)d * 256 + h * 128 + 4 * q;
    atomicAdd(ap + 0, hv.x * alpha);
    atomicAdd(ap + 1, hv.y * alpha);
    atomicAdd(ap + 2, hv.z * alpha);
    atomicAdd(ap + 3, hv.w * alpha);
  }
}

// ------------------------------------------- head-mean + bias + sigmoid
__global__ void finalize_kernel(const float* __restrict__ acc, const float* __restrict__ bias,
                                _Float16* __restrict__ out16, float* __restrict__ out32,
                                int n) {
  int i = blockIdx.x * blockDim.x + threadIdx.x;
  if (i >= n * 128) return;
  int node = i >> 7, c = i & 127;
  float v = 0.5f * (acc[(size_t)node * 256 + c] + acc[(size_t)node * 256 + 128 + c]) + bias[c];
  v = sigmoidf_(v);
  out16[i] = (_Float16)v;
  if (out32) out32[i] = v;
}

// ------------------------------------------- pooling: gate logits (BN+relu+dot)
__global__ void pool_gate_kernel(const float* __restrict__ gh, const float* __restrict__ gb1,
                                 const float* __restrict__ bg, const float* __restrict__ bb,
                                 const float* __restrict__ bm, const float* __restrict__ bv,
                                 const float* __restrict__ gw2, const float* __restrict__ gb2,
                                 const int* __restrict__ batch,
                                 float* __restrict__ logits, unsigned* __restrict__ gmax, int n) {
  int w = blockIdx.x * (blockDim.x >> 5) + (threadIdx.x >> 5);
  if (w >= n) return;
  int lane = threadIdx.x & 31;
  float s = 0.f;
#pragma unroll
  for (int i = 0; i < 4; ++i) {
    int c = lane + 32 * i;
    float v = gh[(size_t)w * 128 + c] + gb1[c];
    v = (v - bm[c]) * rsqrtf(bv[c] + 1e-5f) * bg[c] + bb[c];
    v = v > 0.f ? v : 0.f;
    s += v * gw2[c];
  }
#pragma unroll
  for (int o = 16; o; o >>= 1) s += __shfl_xor(s, o, 32);
  if (lane == 0) {
    s += gb2[0];
    logits[w] = s;
    atomicMax(&gmax[batch[w]], fflip(s));
  }
}

__global__ void pool_exp_kernel(const float* __restrict__ logits, const unsigned* __restrict__ gmax,
                                const int* __restrict__ batch, float* __restrict__ evb,
                                float* __restrict__ gsum, int n) {
  int i = blockIdx.x * blockDim.x + threadIdx.x;
  if (i >= n) return;
  int b = batch[i];
  float ev = expf(logits[i] - funflip(gmax[b]));
  evb[i] = ev;
  atomicAdd(&gsum[b], ev);
}

__global__ void pool_scatter_kernel(const float* __restrict__ evb, const float* __restrict__ gsum,
                                    const int* __restrict__ batch, const float* __restrict__ out2,
                                    float* __restrict__ pooled, int n) {
  int i = blockIdx.x * blockDim.x + threadIdx.x;
  if (i >= n * 128) return;
  int node = i >> 7, c = i & 127;
  int b = batch[node];
  float gate = evb[node] / (gsum[b] + 1e-16f);
  atomicAdd(&pooled[(size_t)b * 128 + c], gate * out2[i]);
}

__global__ void pool_out_kernel(const float* __restrict__ pooled, const float* __restrict__ glw,
                                const float* __restrict__ glb, float* __restrict__ out, int G) {
  int w = blockIdx.x * (blockDim.x >> 5) + (threadIdx.x >> 5);
  if (w >= G) return;
  int lane = threadIdx.x & 31;
  float s = 0.f;
#pragma unroll
  for (int i = 0; i < 4; ++i) {
    int c = lane + 32 * i;
    s += pooled[(size_t)w * 128 + c] * glw[c];
  }
#pragma unroll
  for (int o = 16; o; o >>= 1) s += __shfl_xor(s, o, 32);
  if (lane == 0) out[w] = sigmoidf_(s + glb[0]);
}

// ================================================================ host
static inline int cdiv(long a, int b) { return (int)((a + b - 1) / b); }

extern "C" void kernel_launch(void* const* d_in, const int* in_sizes, int n_in,
                              void* d_out, int out_size, void* d_ws, size_t ws_size,
                              hipStream_t stream) {
  const int* xi    = (const int*)d_in[0];
  const int* ei    = (const int*)d_in[1];
  const int* batch = (const int*)d_in[3];
  const float* emb[6];
  for (int i = 0; i < 6; ++i) emb[i] = (const float*)d_in[4 + i];
  const float* W1  = (const float*)d_in[10];
  const float* as1 = (const float*)d_in[11];
  const float* ad1 = (const float*)d_in[12];
  const float* b1  = (const float*)d_in[13];
  const float* W2  = (const float*)d_in[14];
  const float* as2 = (const float*)d_in[15];
  const float* ad2 = (const float*)d_in[16];
  const float* b2  = (const float*)d_in[17];
  const float* gw1 = (const float*)d_in[18];
  const float* gb1 = (const float*)d_in[19];
  const float* bng = (const float*)d_in[20];
  const float* bnb = (const float*)d_in[21];
  const float* bnm = (const float*)d_in[22];
  const float* bnv = (const float*)d_in[23];
  const float* gw2 = (const float*)d_in[24];
  const float* gb2 = (const float*)d_in[25];
  const float* glw = (const float*)d_in[26];
  const float* glb = (const float*)d_in[27];

  const int E = in_sizes[2];       // edge_type count == E
  const int N = in_sizes[3];       // batch count == N
  const int G = out_size;
  const int ET = E + N;            // edges + self loops
  const int* src = ei;
  const int* dst = ei + E;

  // ---- workspace carve-up (256B aligned)
  size_t off = 0;
  char* base = (char*)d_ws;
  auto alloc = [&](size_t bytes) {
    void* p = base + off;
    off += (bytes + 255) & ~(size_t)255;
    return p;
  };
  _Float16* a16   = (_Float16*)alloc((size_t)N * 384 * 2); // xcat f16, reused as out1/out2 f16
  float*    hbuf  = (float*)alloc((size_t)N * 256 * 4);    // GEMM out (h), reused for gh
  float*    accb  = (float*)alloc((size_t)N * 256 * 4);    // segment-sum accumulator
  float*    out2  = (float*)alloc((size_t)N * 128 * 4);    // layer-2 output f32
  float*    als   = (float*)alloc((size_t)N * 2 * 4);
  float*    ald   = (float*)alloc((size_t)N * 2 * 4);
  unsigned* maxb  = (unsigned*)alloc((size_t)N * 2 * 4);
  float*    sumb  = (float*)alloc((size_t)N * 2 * 4);
  float*    ew    = (float*)alloc((size_t)ET * 2 * 4);
  unsigned* wp1   = (unsigned*)alloc((size_t)192 * 256 * 4);
  unsigned* wp2   = (unsigned*)alloc((size_t)64 * 256 * 4);
  unsigned* wp3   = (unsigned*)alloc((size_t)64 * 128 * 4);
  float*    logit = (float*)alloc((size_t)N * 4);
  float*    evb   = (float*)alloc((size_t)N * 4);
  unsigned* gmax  = (unsigned*)alloc((size_t)G * 4);
  float*    gsum  = (float*)alloc((size_t)G * 4);
  float*    pool  = (float*)alloc((size_t)G * 128 * 4);
  (void)ws_size; (void)n_in;

  const int B = 256;
  const unsigned NEG_INF_FLIP = 0x007FFFFFu; // fflip(-inf)

  // ---- weight packing (f16 WMMA fragment-major layout)
  pack_w_frag_kernel<<<cdiv(192L * 256, B), B, 0, stream>>>(W1, wp1, 384, 256);
  pack_w_frag_kernel<<<cdiv(64L * 256, B), B, 0, stream>>>(W2, wp2, 128, 256);
  pack_w_frag_kernel<<<cdiv(64L * 128, B), B, 0, stream>>>(gw1, wp3, 128, 128);

  // ---- embedding gather
  embed_kernel<<<cdiv((long)N * 384, B), B, 0, stream>>>(
      xi, emb[0], emb[1], emb[2], emb[3], emb[4], emb[5], a16, N);

  const int rt = (N + 15) / 16; // row tiles

  // ================= GAT layer 1 =================
  gemm_wmma_kernel<384, 256><<<dim3(rt), 128, 0, stream>>>(
      (const unsigned*)a16, (const uint4*)wp1, hbuf, N);
  attn_logits_kernel<<<cdiv((long)N * 2, 8), 256, 0, stream>>>(hbuf, as1, ad1, als, ald, N);
  fill_u32_kernel<<<1024, B, 0, stream>>>(maxb, NEG_INF_FLIP, (size_t)N * 2);
  fill_u32_kernel<<<1024, B, 0, stream>>>((unsigned*)sumb, 0u, (size_t)N * 2);
  fill_u32_kernel<<<4096, B, 0, stream>>>((unsigned*)accb, 0u, (size_t)N * 256);
  edge_max_kernel<<<cdiv(ET, B), B, 0, stream>>>(src, dst, als, ald, ew, maxb, E, ET);
  edge_exp_kernel<<<cdiv(ET, B), B, 0, stream>>>(dst, ew, maxb, sumb, E, ET);
  edge_scatter_kernel<<<cdiv((long)ET * 32, B), B, 0, stream>>>(
      src, dst, ew, sumb, hbuf, accb, E, ET);
  finalize_kernel<<<cdiv((long)N * 128, B), B, 0, stream>>>(accb, b1, a16, (float*)nullptr, N);

  // ================= GAT layer 2 =================
  gemm_wmma_kernel<128, 256><<<dim3(rt), 128, 0, stream>>>(
      (const unsigned*)a16, (const uint4*)wp2, hbuf, N);
  attn_logits_kernel<<<cdiv((long)N * 2, 8), 256, 0, stream>>>(hbuf, as2, ad2, als, ald, N);
  fill_u32_kernel<<<1024, B, 0, stream>>>(maxb, NEG_INF_FLIP, (size_t)N * 2);
  fill_u32_kernel<<<1024, B, 0, stream>>>((unsigned*)sumb, 0u, (size_t)N * 2);
  fill_u32_kernel<<<4096, B, 0, stream>>>((unsigned*)accb, 0u, (size_t)N * 256);
  edge_max_kernel<<<cdiv(ET, B), B, 0, stream>>>(src, dst, als, ald, ew, maxb, E, ET);
  edge_exp_kernel<<<cdiv(ET, B), B, 0, stream>>>(dst, ew, maxb, sumb, E, ET);
  edge_scatter_kernel<<<cdiv((long)ET * 32, B), B, 0, stream>>>(
      src, dst, ew, sumb, hbuf, accb, E, ET);
  finalize_kernel<<<cdiv((long)N * 128, B), B, 0, stream>>>(accb, b2, a16, out2, N);

  // ================= GlobalAttention pooling =================
  gemm_wmma_kernel<128, 128><<<dim3(rt), 128, 0, stream>>>(
      (const unsigned*)a16, (const uint4*)wp3, hbuf, N); // gh (pre-bias) into hbuf
  fill_u32_kernel<<<4, B, 0, stream>>>(gmax, NEG_INF_FLIP, (size_t)G);
  fill_u32_kernel<<<4, B, 0, stream>>>((unsigned*)gsum, 0u, (size_t)G);
  fill_u32_kernel<<<64, B, 0, stream>>>((unsigned*)pool, 0u, (size_t)G * 128);
  pool_gate_kernel<<<cdiv(N, 8), 256, 0, stream>>>(
      hbuf, gb1, bng, bnb, bnm, bnv, gw2, gb2, batch, logit, gmax, N);
  pool_exp_kernel<<<cdiv(N, B), B, 0, stream>>>(logit, gmax, batch, evb, gsum, N);
  pool_scatter_kernel<<<cdiv((long)N * 128, B), B, 0, stream>>>(evb, gsum, batch, out2, pool, N);
  pool_out_kernel<<<cdiv(G, 8), 256, 0, stream>>>(pool, glw, glb, (float*)d_out, G);
}
